// VQVAE_3779571221105
// MI455X (gfx1250) — matrix operand order repaired and checked by reading, not compiled, not run on hardware
//
#include <hip/hip_runtime.h>
#include <math.h>

typedef _Float16 half_t;
typedef __attribute__((ext_vector_type(16))) _Float16 v16h;
typedef __attribute__((ext_vector_type(8)))  _Float16 v8h;
typedef __attribute__((ext_vector_type(8)))  float    v8f;

// ---------------- small helpers ----------------
__device__ __forceinline__ v8f zerof8() {
    v8f r;
#pragma unroll
    for (int i = 0; i < 8; ++i) r[i] = 0.0f;
    return r;
}
__device__ __forceinline__ v8h ld8(const half_t* p) { return *(const v8h*)p; }
__device__ __forceinline__ v16h cat16(v8h lo, v8h hi) {
    v16h r;
#pragma unroll
    for (int i = 0; i < 8; ++i) { r[i] = lo[i]; r[i + 8] = hi[i]; }
    return r;
}
__device__ __forceinline__ v8f wmma_f16(v16h a, v16h b, v8f c) {
    return __builtin_amdgcn_wmma_f32_16x16x32_f16(false, a, false, b, (short)0, c,
                                                  false, false);
}

// ---------------- halo zero for padded NHWC activations ----------------
__global__ void halo_zero(half_t* __restrict__ buf, int Hp, int Wp) {
    const int k = blockIdx.x, img = blockIdx.y, c = threadIdx.x;
    int ih, iw;
    if (k < Wp)            { ih = 0;      iw = k; }
    else if (k < 2 * Wp)   { ih = Hp - 1; iw = k - Wp; }
    else {
        int k2 = k - 2 * Wp;
        ih = 1 + (k2 >> 1);
        iw = (k2 & 1) ? (Wp - 1) : 0;
    }
    buf[((((size_t)img * Hp + ih) * Wp) + iw) * 256 + c] = (half_t)0;
}

// ---------------- weight / codebook pre-pack ----------------
__global__ void prep_w1(const float* __restrict__ We1, half_t* __restrict__ W1p) {
    int id = blockIdx.x * 256 + threadIdx.x;        // 12288
    W1p[id] = (half_t)We1[id];
}
__global__ void prep_w2(const float* __restrict__ We2, half_t* __restrict__ Wp2) {
    int id = blockIdx.x * 256 + threadIdx.x;        // 1048576
    int o = id >> 12, tap = (id >> 8) & 15, ic = id & 255;
    Wp2[id] = (half_t)We2[o * 4096 + ic * 16 + tap];
}
__global__ void prep_wd1(const float* __restrict__ Wd1, half_t* __restrict__ Bd1) {
    int id = blockIdx.x * 256 + threadIdx.x;        // 1048576
    int o = id >> 12, tap = (id >> 8) & 15, ic = id & 255;
    Bd1[id] = (half_t)Wd1[(tap * 256 + ic) * 256 + o];
}
__global__ void prep_wd2(const float* __restrict__ Wd2, half_t* __restrict__ Bd2) {
    int id = blockIdx.x * 256 + threadIdx.x;        // 65536
    int o = id >> 12, tap = (id >> 8) & 15, ic = id & 255;
    Bd2[id] = (o < 3) ? (half_t)Wd2[(tap * 256 + ic) * 3 + o] : (half_t)0;
}
__global__ __launch_bounds__(64) void prep_cb(const float* __restrict__ cbf,
                                              half_t* __restrict__ cb16,
                                              float* __restrict__ cnorm) {
    __shared__ float red[64];
    int n = blockIdx.x, tid = threadIdx.x;
    float s = 0.0f;
    if (n < 1000) {
        for (int j = tid; j < 256; j += 64) {
            float v = cbf[n * 256 + j];
            cb16[n * 256 + j] = (half_t)v;
            s += v * v;
        }
    } else {
        for (int j = tid; j < 256; j += 64) cb16[n * 256 + j] = (half_t)0;
    }
    red[tid] = s;
    __syncthreads();
    if (tid == 0) {
        float t = 0.0f;
        for (int i = 0; i < 64; ++i) t += red[i];
        cnorm[n] = (n < 1000) ? t : 3.0e38f;
    }
}

// ---------------- conv1: 3->256, k4 s2 p1, ReLU -> padded z1p ----------------
__global__ __launch_bounds__(64) void conv1_relu(const float* __restrict__ x,
                                                 const half_t* __restrict__ W1p,
                                                 const float* __restrict__ be1,
                                                 half_t* __restrict__ z1p) {
    __shared__ half_t wlds[64 * 48];
    const int tid = threadIdx.x;
    const int ocb = blockIdx.y * 64;
    for (int t = tid; t < 64 * 48; t += 64) wlds[t] = W1p[ocb * 48 + t];
    __syncthreads();

    const int pix = blockIdx.x * 64 + tid;          // b*16384 + oh*128 + ow
    const int b = pix >> 14, oh = (pix >> 7) & 127, ow = pix & 127;
    float p[48];
#pragma unroll
    for (int c = 0; c < 3; ++c)
#pragma unroll
        for (int kh = 0; kh < 4; ++kh)
#pragma unroll
            for (int kw = 0; kw < 4; ++kw) {
                int ih = 2 * oh - 1 + kh, iw = 2 * ow - 1 + kw;
                bool ok = ((unsigned)ih < 256u) && ((unsigned)iw < 256u);
                p[c * 16 + kh * 4 + kw] =
                    ok ? x[((b * 3 + c) << 16) + (ih << 8) + iw] : 0.0f;
            }
    half_t* orow =
        z1p + (((size_t)(b * 130 + oh + 1) * 130) + ow + 1) * 256 + ocb;
    for (int oc = 0; oc < 64; ++oc) {
        float acc = be1[ocb + oc];
#pragma unroll
        for (int j = 0; j < 48; ++j) acc += p[j] * (float)wlds[oc * 48 + j];
        orow[oc] = (half_t)fmaxf(acc, 0.0f);
    }
}

// ---------------- conv2: 256->256 WMMA, 32x32 register-blocked ---------------
// z1p padded NHWC [8][130][130][256] -> zf f32 [32768][256] + zf16
__global__ __launch_bounds__(256) void conv2_wmma(const half_t* __restrict__ z1p,
                                                  const half_t* __restrict__ Wp2,
                                                  const float* __restrict__ be2,
                                                  float* __restrict__ zf,
                                                  half_t* __restrict__ zf16) {
    const int tid = threadIdx.x, lane = tid & 31, wave = tid >> 5;
    const int m0 = blockIdx.x * 32;                 // 32 rows per block
    const int b = m0 >> 12, oh = (m0 >> 6) & 63;
    const int r = lane & 15;
    const int koff = (lane < 16) ? 0 : 8;
    const int boff = (lane < 16) ? 0 : 8;
    const int n0 = wave * 32;
    const half_t* wrow0 = Wp2 + (size_t)(n0 + r) * 4096 + boff;
    const half_t* wrow1 = Wp2 + (size_t)(n0 + 16 + r) * 4096 + boff;
    // M-tile 0: ow = ow0 + r, M-tile 1: ow = ow0 + 16 + r (A offset +8192 halves)
    const size_t pix0 = (size_t)(b * 130 + 2 * oh) * 130 + 2 * ((m0 & 63) + r);

    v8f acc00 = zerof8(), acc01 = zerof8(), acc10 = zerof8(), acc11 = zerof8();
    for (int kh = 0; kh < 4; ++kh) {
        const half_t* Ab = z1p + (pix0 + (size_t)kh * 130) * 256 + koff;
        const half_t* B0 = wrow0 + kh * 1024;
        const half_t* B1 = wrow1 + kh * 1024;
        __builtin_prefetch(Ab + 130 * 256, 0, 1);
#pragma unroll 4
        for (int cc = 0; cc < 32; ++cc) {
            v16h a0 = cat16(ld8(Ab + cc * 32), ld8(Ab + cc * 32 + 16));
            v16h a1 = cat16(ld8(Ab + cc * 32 + 8192), ld8(Ab + cc * 32 + 8208));
            v16h b0 = cat16(ld8(B0 + cc * 32), ld8(B0 + cc * 32 + 16));
            v16h b1 = cat16(ld8(B1 + cc * 32), ld8(B1 + cc * 32 + 16));
            acc00 = wmma_f16(a0, b0, acc00);
            acc01 = wmma_f16(a0, b1, acc01);
            acc10 = wmma_f16(a1, b0, acc10);
            acc11 = wmma_f16(a1, b1, acc11);
        }
    }
    const float bias0 = be2[n0 + r], bias1 = be2[n0 + 16 + r];
#pragma unroll
    for (int i = 0; i < 8; ++i) {
        const int M = i + ((lane < 16) ? 0 : 8);
        const size_t ma = (size_t)(m0 + M) * 256;
        const size_t mb = (size_t)(m0 + 16 + M) * 256;
        float v00 = acc00[i] + bias0, v01 = acc01[i] + bias1;
        float v10 = acc10[i] + bias0, v11 = acc11[i] + bias1;
        zf[ma + n0 + r] = v00;       zf[ma + n0 + 16 + r] = v01;
        zf[mb + n0 + r] = v10;       zf[mb + n0 + 16 + r] = v11;
        zf16[ma + n0 + r] = (half_t)v00; zf16[ma + n0 + 16 + r] = (half_t)v01;
        zf16[mb + n0 + r] = (half_t)v10; zf16[mb + n0 + 16 + r] = (half_t)v11;
    }
}

// ---------------- VQ argmin: WMMA distances + shfl butterfly -----------------
__global__ __launch_bounds__(256) void vq_argmin(const half_t* __restrict__ zf16,
                                                 const half_t* __restrict__ cb16,
                                                 const float* __restrict__ cnorm,
                                                 int* __restrict__ vidx) {
    const int tid = threadIdx.x, lane = tid & 31, wave = tid >> 5;
    const int m0 = (blockIdx.x * 8 + wave) * 16;
    const int r = lane & 15;
    const int koff = (lane < 16) ? 0 : 8;
    const int boff = (lane < 16) ? 0 : 8;
    const int m = m0 + r;

    v16h A[8];
#pragma unroll
    for (int kc = 0; kc < 8; ++kc) {
        const half_t* ap = zf16 + (size_t)m * 256 + kc * 32 + koff;
        A[kc] = cat16(ld8(ap), ld8(ap + 16));
    }
    float bestv[8];
    int besti[8];
#pragma unroll
    for (int i = 0; i < 8; ++i) { bestv[i] = 3.4e38f; besti[i] = 0; }

    for (int nc = 0; nc < 63; ++nc) {
        const int n = nc * 16 + r;
        const half_t* crow = cb16 + (size_t)n * 256 + boff;
        v8f acc = zerof8();
#pragma unroll
        for (int kc = 0; kc < 8; ++kc) {
            v16h bf = cat16(ld8(crow + kc * 32), ld8(crow + kc * 32 + 16));
            acc = wmma_f16(A[kc], bf, acc);
        }
        const float cn = cnorm[n];
#pragma unroll
        for (int i = 0; i < 8; ++i) {
            float v = cn - 2.0f * acc[i];
            int bi = n;
#pragma unroll
            for (int mm = 1; mm < 16; mm <<= 1) {
                float ov = __shfl_xor(v, mm, 32);
                int oi = __shfl_xor(bi, mm, 32);
                if (ov < v) { v = ov; bi = oi; }
            }
            if (v < bestv[i]) { bestv[i] = v; besti[i] = bi; }
        }
    }
    if ((lane & 15) == 0) {
        const int mbase = m0 + ((lane < 16) ? 0 : 8);
#pragma unroll
        for (int i = 0; i < 8; ++i) vidx[mbase + i] = besti[i];
    }
}

// gather e = codebook[idx] -> flat f32 out + padded f16 for decoder
__global__ void vq_gather(const int* __restrict__ vidx,
                          const float* __restrict__ cbf,
                          float* __restrict__ eout, half_t* __restrict__ e16p) {
    int id = blockIdx.x * 256 + threadIdx.x;        // 8388608
    int mrow = id >> 8, c = id & 255;
    int k = vidx[mrow];
    float v = cbf[k * 256 + c];
    eout[id] = v;
    const int b = mrow >> 12, h = (mrow >> 6) & 63, w = mrow & 63;
    e16p[(((size_t)(b * 66 + h + 1) * 66) + w + 1) * 256 + c] = (half_t)v;
}

// ---------------- tconv1: 256->256, 2x up, ReLU, 32x32 register-blocked ------
// e16p padded [8][66][66][256] -> y1p padded [8][130][130][256]
__global__ __launch_bounds__(256) void tconv1_wmma(const half_t* __restrict__ e16p,
                                                   const half_t* __restrict__ Bd1,
                                                   const float* __restrict__ bd1,
                                                   half_t* __restrict__ y1p) {
    const int tid = threadIdx.x, lane = tid & 31, wave = tid >> 5;
    const int t = blockIdx.x;                       // 4096
    const int owp = t & 1, pw = (t >> 1) & 1, oh = (t >> 2) & 127, b = t >> 9;
    const int owt0 = owp * 2;                       // tiles owt0, owt0+1
    const int ph = oh & 1;
    const int r = lane & 15;
    const int koff = (lane < 16) ? 0 : 8;
    const int boff = (lane < 16) ? 0 : 8;
    const int n0 = wave * 32;
    const half_t* wrow0 = Bd1 + (size_t)(n0 + r) * 4096 + boff;
    const half_t* wrow1 = Bd1 + (size_t)(n0 + 16 + r) * 4096 + boff;
    const int ihb1 = (oh >> 1) + ph + 1;
    const int iwb1 = owt0 * 16 + r + pw + 1;        // tile 1 at +16 px (+4096 h)

    v8f acc00 = zerof8(), acc01 = zerof8(), acc10 = zerof8(), acc11 = zerof8();
#pragma unroll
    for (int tap = 0; tap < 4; ++tap) {
        const int th = tap >> 1, tw = tap & 1;
        const int kh = 2 * th + 1 - ph, kw = 2 * tw + 1 - pw;
        const half_t* Ab =
            e16p + (((size_t)(b * 66 + ihb1 - th) * 66) + iwb1 - tw) * 256 + koff;
        const int wo = (kh * 4 + kw) << 8;
#pragma unroll
        for (int cc = 0; cc < 8; ++cc) {
            v16h a0 = cat16(ld8(Ab + cc * 32), ld8(Ab + cc * 32 + 16));
            v16h a1 = cat16(ld8(Ab + cc * 32 + 4096), ld8(Ab + cc * 32 + 4112));
            v16h b0 = cat16(ld8(wrow0 + wo + cc * 32), ld8(wrow0 + wo + cc * 32 + 16));
            v16h b1 = cat16(ld8(wrow1 + wo + cc * 32), ld8(wrow1 + wo + cc * 32 + 16));
            acc00 = wmma_f16(a0, b0, acc00);
            acc01 = wmma_f16(a0, b1, acc01);
            acc10 = wmma_f16(a1, b0, acc10);
            acc11 = wmma_f16(a1, b1, acc11);
        }
    }
    const float bias0 = bd1[n0 + r], bias1 = bd1[n0 + 16 + r];
#pragma unroll
    for (int i = 0; i < 8; ++i) {
        const int M = i + ((lane < 16) ? 0 : 8);
        const int owM0 = pw + 2 * (owt0 * 16 + M);  // tile 1: +32
        half_t* row0 =
            y1p + (((size_t)(b * 130 + oh + 1) * 130) + owM0 + 1) * 256;
        half_t* row1 = row0 + 32 * 256;
        row0[n0 + r]      = (half_t)fmaxf(acc00[i] + bias0, 0.0f);
        row0[n0 + 16 + r] = (half_t)fmaxf(acc01[i] + bias1, 0.0f);
        row1[n0 + r]      = (half_t)fmaxf(acc10[i] + bias0, 0.0f);
        row1[n0 + 16 + r] = (half_t)fmaxf(acc11[i] + bias1, 0.0f);
    }
}

// ---------------- tconv2: 256->3 (N pad 16), 2x up, tanh, 2 M-tiles/wave -----
// y1p padded [8][130][130][256] -> x_hat NCHW f32 [8][3][256][256]
__global__ __launch_bounds__(256) void tconv2_wmma(const half_t* __restrict__ y1p,
                                                   const half_t* __restrict__ Bd2,
                                                   const float* __restrict__ bd2,
                                                   float* __restrict__ xhat) {
    const int tid = threadIdx.x, lane = tid & 31, wave = tid >> 5;
    const int t = blockIdx.x * 8 + wave;            // 16384 tile-pairs
    const int owp = t & 3, pw = (t >> 2) & 1, oh = (t >> 3) & 255, b = t >> 11;
    const int owt0 = owp * 2;
    const int ph = oh & 1;
    const int r = lane & 15;
    const int koff = (lane < 16) ? 0 : 8;
    const int boff = (lane < 16) ? 0 : 8;
    const half_t* wrow = Bd2 + (size_t)r * 4096 + boff;
    const int ihb1 = (oh >> 1) + ph + 1;
    const int iwb1 = owt0 * 16 + r + pw + 1;

    v8f acc0 = zerof8(), acc1 = zerof8();
#pragma unroll
    for (int tap = 0; tap < 4; ++tap) {
        const int th = tap >> 1, tw = tap & 1;
        const int kh = 2 * th + 1 - ph, kw = 2 * tw + 1 - pw;
        const half_t* Ab =
            y1p + (((size_t)(b * 130 + ihb1 - th) * 130) + iwb1 - tw) * 256 + koff;
        const int wo = (kh * 4 + kw) << 8;
#pragma unroll
        for (int cc = 0; cc < 8; ++cc) {
            v16h a0 = cat16(ld8(Ab + cc * 32), ld8(Ab + cc * 32 + 16));
            v16h a1 = cat16(ld8(Ab + cc * 32 + 4096), ld8(Ab + cc * 32 + 4112));
            v16h bf = cat16(ld8(wrow + wo + cc * 32), ld8(wrow + wo + cc * 32 + 16));
            acc0 = wmma_f16(a0, bf, acc0);
            acc1 = wmma_f16(a1, bf, acc1);
        }
    }
    if (r < 3) {
        const float bias = bd2[r];
#pragma unroll
        for (int i = 0; i < 8; ++i) {
            const int M = i + ((lane < 16) ? 0 : 8);
            const int owM0 = pw + 2 * (owt0 * 16 + M);
            float* orow = xhat + (((size_t)(b * 3 + r) * 256 + oh) << 8);
            orow[owM0]      = tanhf(acc0[i] + bias);
            orow[owM0 + 32] = tanhf(acc1[i] + bias);
        }
    }
}

// ---------------- host launcher ----------------
extern "C" void kernel_launch(void* const* d_in, const int* in_sizes, int n_in,
                              void* d_out, int out_size, void* d_ws, size_t ws_size,
                              hipStream_t stream) {
    (void)in_sizes; (void)n_in; (void)out_size; (void)ws_size;
    const float* x   = (const float*)d_in[0];
    const float* cbf = (const float*)d_in[1];
    const float* We1 = (const float*)d_in[2];
    const float* be1 = (const float*)d_in[3];
    const float* We2 = (const float*)d_in[4];
    const float* be2 = (const float*)d_in[5];
    const float* Wd1 = (const float*)d_in[6];
    const float* bd1 = (const float*)d_in[7];
    const float* Wd2 = (const float*)d_in[8];
    const float* bd2 = (const float*)d_in[9];

    float* out  = (float*)d_out;
    float* xhat = out;                          // 8*3*256*256   = 1572864
    float* zf   = out + 1572864;                // 32768*256     = 8388608
    float* eout = out + 1572864 + 8388608;      // 32768*256

    char* ws = (char*)d_ws;
    size_t off = 0;
    auto take = [&](size_t bytes) -> char* {
        char* p = ws + off;
        off = (off + bytes + 255) & ~(size_t)255;
        return p;
    };
    half_t* z1p   = (half_t*)take((size_t)8 * 130 * 130 * 256 * 2);
    half_t* y1p   = (half_t*)take((size_t)8 * 130 * 130 * 256 * 2);
    half_t* e16p  = (half_t*)take((size_t)8 * 66 * 66 * 256 * 2);
    half_t* zf16  = (half_t*)take(8388608ull * 2);
    half_t* Wp2   = (half_t*)take(1048576ull * 2);
    half_t* Bd1p  = (half_t*)take(1048576ull * 2);
    half_t* Bd2p  = (half_t*)take(65536ull * 2);
    half_t* cb16  = (half_t*)take(258048ull * 2);
    float*  cnorm = (float*)take(1008ull * 4);
    half_t* W1p   = (half_t*)take(12288ull * 2);
    int*    vidx  = (int*)take(32768ull * 4);

    halo_zero<<<dim3(516, 8), 256, 0, stream>>>(z1p, 130, 130);
    halo_zero<<<dim3(516, 8), 256, 0, stream>>>(y1p, 130, 130);
    halo_zero<<<dim3(260, 8), 256, 0, stream>>>(e16p, 66, 66);
    prep_w1 <<<48,   256, 0, stream>>>(We1, W1p);
    prep_w2 <<<4096, 256, 0, stream>>>(We2, Wp2);
    prep_wd1<<<4096, 256, 0, stream>>>(Wd1, Bd1p);
    prep_wd2<<<256,  256, 0, stream>>>(Wd2, Bd2p);
    prep_cb <<<1008, 64,  0, stream>>>(cbf, cb16, cnorm);

    conv1_relu <<<dim3(2048, 4), 64, 0, stream>>>(x, W1p, be1, z1p);
    conv2_wmma <<<1024,  256, 0, stream>>>(z1p, Wp2, be2, zf, zf16);
    vq_argmin  <<<256,   256, 0, stream>>>(zf16, cb16, cnorm, vidx);
    vq_gather  <<<32768, 256, 0, stream>>>(vidx, cbf, eout, e16p);
    tconv1_wmma<<<4096,  256, 0, stream>>>(e16p, Bd1p, bd1, y1p);
    tconv2_wmma<<<2048,  256, 0, stream>>>(y1p, Bd2p, bd2, xhat);
}
